// TemporalGraphNetwork_63299228009046
// MI455X (gfx1250) — matrix operand order repaired and checked by reading, not compiled, not run on hardware
//
#include <hip/hip_runtime.h>

// ---------------------------------------------------------------------------
// Temporal Graph Network forward, fused for MI455X (gfx1250, wave32, WMMA).
// All GEMMs run through v_wmma_f32_16x16x32_bf16 (f32 accumulate).
// edge_emb (205 MB fp32) is never materialized in HBM: the edge MLP, the
// node-embedding gathers and the message MLP are fused per 32-edge tile.
// node_emb is kept as a bf16 table (12.8 MB) in workspace -> L2 resident.
// Gathers of node_emb rows use GLOBAL_LOAD_ASYNC_TO_LDS_B128 (ASYNCcnt path),
// bypassing VGPRs and overlapping with the fp32->bf16 edge-feature staging.
// ---------------------------------------------------------------------------

typedef __bf16 bf16;
typedef bf16  v16bf __attribute__((ext_vector_type(16)));
typedef bf16  v8bf  __attribute__((ext_vector_type(8)));
typedef float v8f   __attribute__((ext_vector_type(8)));

static constexpr int kNodes   = 50000;
static constexpr int kEdges   = 400000;
static constexpr int kNodeDim = 128;
static constexpr int kEdgeDim = 64;
static constexpr int kHidden  = 128;
static constexpr int kTile    = 32;    // rows (nodes/edges) per workgroup
static constexpr int kThreads = 128;   // 4 wave32s per workgroup

// ---------------------------------------------------------------------------
// CDNA5 async memory->LDS copy (no VGPR roundtrip, tracked by ASYNCcnt).
// The low 32 bits of a generic pointer to __shared__ are the workgroup-
// relative LDS byte offset (LDS aperture maps addr[31:0] directly).
// ---------------------------------------------------------------------------
__device__ __forceinline__ void async_copy_b128(void* lds_dst,
                                                const void* gsrc) {
  const unsigned lds = (unsigned)(uintptr_t)lds_dst;
  asm volatile("global_load_async_to_lds_b128 %0, %1, off"
               :
               : "v"(lds), "v"(gsrc)
               : "memory");
}
__device__ __forceinline__ void wait_asynccnt_zero() {
  asm volatile("s_wait_asynccnt 0x0" ::: "memory");
}

// ---------------------------------------------------------------------------
// WMMA fragment loaders (layouts per CDNA5 ISA 7.12.2, wave32).
// A (16x32 bf16): lane L holds row M = L%16; K runs [half*8 .. half*8+7] and
//                 [16+half*8 .. 16+half*8+7]  (half = L/16), 2 bf16 per VGPR.
// B (32x16 bf16): lane L holds col N = L%16; K runs [half*8 .. half*8+7] and
//                 [16+half*8 .. 16+half*8+7] in the transposed weight Wt[N][K].
// C/D (16x16 f32): element v of lane L is row m0 + (L/16)*8 + v, col n0+L%16.
// ---------------------------------------------------------------------------
__device__ __forceinline__ v16bf load_a_frag(const bf16* A, int lda,
                                             int m0, int k0, int lane) {
  const int row  = m0 + (lane & 15);
  const int half = lane >> 4;
  const v8bf lo = *(const v8bf*)(A + row * lda + k0 + half * 8);
  const v8bf hi = *(const v8bf*)(A + row * lda + k0 + 16 + half * 8);
  return __builtin_shufflevector(lo, hi, 0, 1, 2, 3, 4, 5, 6, 7,
                                 8, 9, 10, 11, 12, 13, 14, 15);
}

__device__ __forceinline__ v16bf load_b_frag(const bf16* Wt, int K,
                                             int n0, int k0, int lane) {
  const int n  = n0 + (lane & 15);
  const int kb = k0 + (lane >> 4) * 8;
  const v8bf lo = *(const v8bf*)(Wt + n * K + kb);
  const v8bf hi = *(const v8bf*)(Wt + n * K + kb + 16);
  return __builtin_shufflevector(lo, hi, 0, 1, 2, 3, 4, 5, 6, 7,
                                 8, 9, 10, 11, 12, 13, 14, 15);
}

// One wave computes a 16-row x 64-col strip (4 accumulators of 16x16).
// Wt is the bf16-transposed weight [128][K]; bias (if INIT) is f32[128].
template <int K, bool INIT>
__device__ __forceinline__ void wmma_gemm(const bf16* A, int lda,
                                          const bf16* Wt, const float* bias,
                                          v8f acc[4], int m0, int n0, int lane) {
  if (INIT) {
#pragma unroll
    for (int j = 0; j < 4; ++j) {
      const float bj = bias[n0 + j * 16 + (lane & 15)];
#pragma unroll
      for (int v = 0; v < 8; ++v) acc[j][v] = bj;
    }
  }
#pragma unroll
  for (int k0 = 0; k0 < K; k0 += 32) {
    const v16bf a = load_a_frag(A, lda, m0, k0, lane);
#pragma unroll
    for (int j = 0; j < 4; ++j) {
      const v16bf b = load_b_frag(Wt, K, n0 + j * 16, k0, lane);
      acc[j] = __builtin_amdgcn_wmma_f32_16x16x32_bf16(
          false, a, false, b, (short)0, acc[j], false, false);
    }
  }
}

__device__ __forceinline__ void store_acc_lds(const v8f acc[4], bf16* out,
                                              int ldo, int m0, int n0,
                                              int lane, bool relu) {
  const int half = lane >> 4, c = lane & 15;
#pragma unroll
  for (int j = 0; j < 4; ++j) {
    const int col = n0 + j * 16 + c;
#pragma unroll
    for (int v = 0; v < 8; ++v) {
      float f = acc[j][v];
      if (relu) f = fmaxf(f, 0.0f);
      out[(m0 + half * 8 + v) * ldo + col] = (bf16)f;
    }
  }
}

// ---------------------------------------------------------------------------
// Weight prep: W (K x H, fp32 row-major) -> Wt (H x K, bf16)
// ---------------------------------------------------------------------------
__global__ void transpose_to_bf16(const float* __restrict__ W,
                                  bf16* __restrict__ out, int K, int H) {
  const int idx = blockIdx.x * 256 + threadIdx.x;
  if (idx < K * H) {
    const int n = idx / K, k = idx % K;
    out[idx] = (bf16)W[k * H + n];
  }
}

// ---------------------------------------------------------------------------
// Node MLP: node_emb = relu(X @ nW1 + nb1) @ nW2 + nb2   (bf16 output table)
// ---------------------------------------------------------------------------
__global__ __launch_bounds__(kThreads) void tgn_node_mlp(
    const float* __restrict__ X, const bf16* __restrict__ nW1t,
    const float* __restrict__ nb1, const bf16* __restrict__ nW2t,
    const float* __restrict__ nb2, bf16* __restrict__ node_emb) {
  __shared__ __align__(32) bf16 sX[kTile * kNodeDim];
  __shared__ __align__(32) bf16 sH[kTile * kHidden];

  const int tid = threadIdx.x, lane = tid & 31, w = tid >> 5;
  const int r0 = blockIdx.x * kTile;

  // Stage X tile (32 x 128 fp32) -> bf16 LDS, zero-pad past kNodes.
  for (int idx = tid; idx < kTile * (kNodeDim / 4); idx += kThreads) {
    const int r = idx >> 5, c = (idx & 31) * 4;
    const int row = r0 + r;
    float x0 = 0.f, x1 = 0.f, x2 = 0.f, x3 = 0.f;
    if (row < kNodes) {
      const float4 vv = *(const float4*)(X + (size_t)row * kNodeDim + c);
      x0 = vv.x; x1 = vv.y; x2 = vv.z; x3 = vv.w;
    }
    bf16* dst = sX + r * kNodeDim + c;
    dst[0] = (bf16)x0; dst[1] = (bf16)x1; dst[2] = (bf16)x2; dst[3] = (bf16)x3;
  }
  __syncthreads();

  const int m0 = (w & 1) * 16, n0 = (w >> 1) * 64;
  v8f acc[4];

  wmma_gemm<kNodeDim, true>(sX, kNodeDim, nW1t, nb1, acc, m0, n0, lane);
  store_acc_lds(acc, sH, kHidden, m0, n0, lane, /*relu=*/true);
  __syncthreads();

  wmma_gemm<kHidden, true>(sH, kHidden, nW2t, nb2, acc, m0, n0, lane);

  const int half = lane >> 4, c = lane & 15;
#pragma unroll
  for (int j = 0; j < 4; ++j) {
    const int col = n0 + j * 16 + c;
#pragma unroll
    for (int v = 0; v < 8; ++v) {
      const int row = r0 + m0 + half * 8 + v;
      if (row < kNodes)
        node_emb[(size_t)row * kHidden + col] = (bf16)acc[j][v];
    }
  }
}

// ---------------------------------------------------------------------------
// Fused edge MLP + gather + message MLP, 32 edges per workgroup.
// out[e] = relu( ne[src] Wa + ee Wb + ne[dst] Wc + mb1 ) @ mW2 + mb2
// where ee = relu(E @ eW1 + eb1) @ eW2 + eb2 stays entirely in LDS.
// ---------------------------------------------------------------------------
__global__ __launch_bounds__(kThreads) void tgn_edge_fused(
    const float* __restrict__ E, const int* __restrict__ edge_index,
    const bf16* __restrict__ node_emb,
    const bf16* __restrict__ eW1t, const float* __restrict__ eb1,
    const bf16* __restrict__ eW2t, const float* __restrict__ eb2,
    const bf16* __restrict__ Wat, const bf16* __restrict__ Wbt,
    const bf16* __restrict__ Wct, const float* __restrict__ mb1,
    const bf16* __restrict__ mW2t, const float* __restrict__ mb2,
    float* __restrict__ out) {
  __shared__ __align__(32) bf16 sE [kTile * kEdgeDim];   //  4 KB
  __shared__ __align__(32) bf16 sH [kTile * kHidden];    //  8 KB (h1, then msg)
  __shared__ __align__(32) bf16 sEE[kTile * kHidden];    //  8 KB (edge_emb)
  __shared__ __align__(32) bf16 sNS[kTile * kHidden];    //  8 KB (ne[src])
  __shared__ __align__(32) bf16 sND[kTile * kHidden];    //  8 KB (ne[dst])

  const int tid = threadIdx.x, lane = tid & 31, w = tid >> 5;
  const int e0 = blockIdx.x * kTile;

  // Prefetch next tile of edge features into L2 (gfx1250 global_prefetch).
  if (blockIdx.x + 1 < gridDim.x) {
    const char* nxt = (const char*)(E + (size_t)(e0 + kTile) * kEdgeDim);
    __builtin_prefetch(nxt + tid * 64, 0, 1);
  }

  // Kick off node-embedding gathers: async memory->LDS (bf16 rows, L2-hit).
  const int* srcI = edge_index;
  const int* dstI = edge_index + kEdges;
  for (int idx = tid; idx < kTile * (kHidden / 8); idx += kThreads) {
    const int r = idx >> 4, c = (idx & 15) * 8;          // 16B chunks
    const int s = srcI[e0 + r];
    const int d = dstI[e0 + r];
    async_copy_b128(sNS + r * kHidden + c,
                    node_emb + (size_t)s * kHidden + c);
    async_copy_b128(sND + r * kHidden + c,
                    node_emb + (size_t)d * kHidden + c);
  }

  // Meanwhile stage the edge-feature tile (32 x 64 fp32) -> bf16 LDS.
  for (int idx = tid; idx < kTile * (kEdgeDim / 4); idx += kThreads) {
    const int r = idx >> 4, c = (idx & 15) * 4;
    const float4 vv = *(const float4*)(E + (size_t)(e0 + r) * kEdgeDim + c);
    bf16* dst = sE + r * kEdgeDim + c;
    dst[0] = (bf16)vv.x; dst[1] = (bf16)vv.y;
    dst[2] = (bf16)vv.z; dst[3] = (bf16)vv.w;
  }

  wait_asynccnt_zero();   // gathers landed in LDS
  __syncthreads();

  const int m0 = (w & 1) * 16, n0 = (w >> 1) * 64;
  v8f acc[4];

  // Edge MLP layer 1: h1 = relu(E @ eW1 + eb1)
  wmma_gemm<kEdgeDim, true>(sE, kEdgeDim, eW1t, eb1, acc, m0, n0, lane);
  store_acc_lds(acc, sH, kHidden, m0, n0, lane, /*relu=*/true);
  __syncthreads();

  // Edge MLP layer 2: ee = h1 @ eW2 + eb2
  wmma_gemm<kHidden, true>(sH, kHidden, eW2t, eb2, acc, m0, n0, lane);
  store_acc_lds(acc, sEE, kHidden, m0, n0, lane, /*relu=*/false);
  __syncthreads();

  // Message: acc = ne[src] Wa + ee Wb + ne[dst] Wc + mb1  (shared accums)
  wmma_gemm<kHidden, true >(sNS, kHidden, Wat, mb1,     acc, m0, n0, lane);
  wmma_gemm<kHidden, false>(sEE, kHidden, Wbt, nullptr, acc, m0, n0, lane);
  wmma_gemm<kHidden, false>(sND, kHidden, Wct, nullptr, acc, m0, n0, lane);
  store_acc_lds(acc, sH, kHidden, m0, n0, lane, /*relu=*/true);  // sH dead->reuse
  __syncthreads();

  // Output projection: out = relu(h) @ mW2 + mb2  (fp32 to HBM)
  wmma_gemm<kHidden, true>(sH, kHidden, mW2t, mb2, acc, m0, n0, lane);

  float* outT = out + (size_t)e0 * kHidden;
  const int half = lane >> 4, c = lane & 15;
#pragma unroll
  for (int j = 0; j < 4; ++j) {
    const int col = n0 + j * 16 + c;
#pragma unroll
    for (int v = 0; v < 8; ++v)
      outT[(size_t)(m0 + half * 8 + v) * kHidden + col] = acc[j][v];
  }
}

// ---------------------------------------------------------------------------
// Host launcher
// ---------------------------------------------------------------------------
extern "C" void kernel_launch(void* const* d_in, const int* in_sizes, int n_in,
                              void* d_out, int out_size, void* d_ws,
                              size_t ws_size, hipStream_t stream) {
  const float* node_features = (const float*)d_in[0];
  const float* edge_features = (const float*)d_in[1];
  /* d_in[2] timestamps: unused by the reference */
  const int*   edge_index    = (const int*)d_in[3];
  const float* nW1 = (const float*)d_in[4];
  const float* nb1 = (const float*)d_in[5];
  const float* nW2 = (const float*)d_in[6];
  const float* nb2 = (const float*)d_in[7];
  const float* eW1 = (const float*)d_in[8];
  const float* eb1 = (const float*)d_in[9];
  const float* eW2 = (const float*)d_in[10];
  const float* eb2 = (const float*)d_in[11];
  const float* mW1 = (const float*)d_in[12];
  const float* mb1 = (const float*)d_in[13];
  const float* mW2 = (const float*)d_in[14];
  const float* mb2 = (const float*)d_in[15];
  float* out = (float*)d_out;

  // Workspace layout (bf16): transposed weights, then the node_emb table.
  bf16* ws   = (bf16*)d_ws;
  bf16* nW1t = ws;
  bf16* nW2t = nW1t + 128 * 128;
  bf16* eW1t = nW2t + 128 * 128;      // [128][64]
  bf16* eW2t = eW1t + 128 * 64;
  bf16* Wat  = eW2t + 128 * 128;
  bf16* Wbt  = Wat  + 128 * 128;
  bf16* Wct  = Wbt  + 128 * 128;
  bf16* mW2t = Wct  + 128 * 128;
  bf16* nemb = mW2t + 128 * 128;      // [50000][128]

  auto T = [&](const float* W, bf16* o, int K, int H) {
    const int n = K * H;
    transpose_to_bf16<<<(n + 255) / 256, 256, 0, stream>>>(W, o, K, H);
  };
  T(nW1, nW1t, 128, 128);
  T(nW2, nW2t, 128, 128);
  T(eW1, eW1t, 64, 128);
  T(eW2, eW2t, 128, 128);
  T(mW1,             Wat, 128, 128);   // mW1 rows   0..127  -> Wa
  T(mW1 + 128 * 128, Wbt, 128, 128);   // mW1 rows 128..255  -> Wb
  T(mW1 + 256 * 128, Wct, 128, 128);   // mW1 rows 256..383  -> Wc
  T(mW2, mW2t, 128, 128);

  const int nodeBlocks = (kNodes + kTile - 1) / kTile;   // 1563
  tgn_node_mlp<<<nodeBlocks, kThreads, 0, stream>>>(
      node_features, nW1t, nb1, nW2t, nb2, nemb);

  const int edgeBlocks = kEdges / kTile;                 // 12500
  tgn_edge_fused<<<edgeBlocks, kThreads, 0, stream>>>(
      edge_features, edge_index, nemb,
      eW1t, eb1, eW2t, eb2, Wat, Wbt, Wct, mb1, mW2t, mb2, out);
}
